// WindowAttention_22204980920601
// MI455X (gfx1250) — matrix operand order repaired and checked by reading, not compiled, not run on hardware
//
#include <hip/hip_runtime.h>
#include <hip/hip_bf16.h>

#define DIM 256
#define WIN 64
#define HEADS 8
#define HEAD_DIM 32
#define NW 64
#define QK_SCALE 0.17677669529663687f  // HEAD_DIM^-0.5

typedef __attribute__((ext_vector_type(16))) __bf16 bf16x16;
typedef __attribute__((ext_vector_type(2)))  __bf16 bf16x2;
typedef __attribute__((ext_vector_type(8)))  float  f32x8;

union WOp { bf16x16 v; unsigned int u[8]; uint4 q[2]; };

__device__ __forceinline__ unsigned short f2bf(float f) {
  unsigned int u = __builtin_bit_cast(unsigned int, f);
  u += 0x7FFFu + ((u >> 16) & 1u);          // round-to-nearest-even
  return (unsigned short)(u >> 16);
}
__device__ __forceinline__ unsigned int pack2(float lo, float hi) {
#if __has_builtin(__builtin_amdgcn_cvt_pk_bf16_f32)
  bf16x2 p = __builtin_amdgcn_cvt_pk_bf16_f32(lo, hi);
  return __builtin_bit_cast(unsigned int, p);
#else
  return (unsigned int)f2bf(lo) | ((unsigned int)f2bf(hi) << 16);
#endif
}

__device__ __forceinline__ f32x8 wmma_bf16(const WOp& a, const WOp& b, f32x8 c) {
  return __builtin_amdgcn_wmma_f32_16x16x32_bf16(
      /*neg_a=*/false, a.v, /*neg_b=*/false, b.v,
      /*c_mod=*/(short)0, c, /*reuse_a=*/false, /*reuse_b=*/false);
}

// A operand (16x32 bf16, MxK). Caller passes absolute row (m) and kh = lane>>4.
// VGPR v holds K pair at kbase + (v>>2)*16 + kh*8 + (v&3)*2.
// Each 4-VGPR half is 16 contiguous bytes -> two b128 loads.
__device__ __forceinline__ void load_a(const unsigned short* base, int row, int stride,
                                       int kbase, int kh, WOp& op) {
  const unsigned short* p = base + row * stride + kbase + kh * 8;
  op.q[0] = *(const uint4*)(p);
  op.q[1] = *(const uint4*)(p + 16);
}

// B operand (32x16 bf16, KxN) from an N-major [n][stride] array (K contiguous).
// Caller passes absolute row (n) and lg = lane>>4.
// VGPR v holds K pair at kbase + (v>>2)*16 + lg*8 + (v&3)*2 -> two b128 loads.
__device__ __forceinline__ void load_b(const unsigned short* base, int row, int stride,
                                       int kbase, int lg, WOp& op) {
  const unsigned short* p = base + row * stride + kbase + lg * 8;
  op.q[0] = *(const uint4*)(p);
  op.q[1] = *(const uint4*)(p + 16);
}

// ---------------------------------------------------------------------------
// Pre-pass: convert fp32 weights to bf16, transposed to N-major for B operands.
// ---------------------------------------------------------------------------
__global__ void convert_weights(const float* __restrict__ w_qkv,
                                const float* __restrict__ w_proj,
                                unsigned short* __restrict__ wqkvT,
                                unsigned short* __restrict__ wprojT) {
  const int tid = blockIdx.x * blockDim.x + threadIdx.x;
  const int stride = gridDim.x * blockDim.x;
  for (int idx = tid; idx < DIM * 3 * DIM; idx += stride) {
    const int k = idx / (3 * DIM), n = idx % (3 * DIM);
    wqkvT[n * DIM + k] = f2bf(w_qkv[idx]);
  }
  for (int idx = tid; idx < DIM * DIM; idx += stride) {
    const int k = idx / DIM, n = idx % DIM;
    wprojT[n * DIM + k] = f2bf(w_proj[idx]);
  }
}

// ---------------------------------------------------------------------------
// Fused window attention: one workgroup (8 wave32) per window.
// LDS: Xs/Os [64][256]bf16 | Qs | Ks | Vt [256][64]bf16 | Ps [8][64][64]bf16 |
//      Bt [127][8]f32 | Ms [64][64]f32   => 224 KB total (<320 KB WGP limit).
// ---------------------------------------------------------------------------
__global__ __launch_bounds__(256)
void winattn_kernel(const float* __restrict__ x, const float* __restrict__ mask,
                    const float* __restrict__ b_qkv, const float* __restrict__ b_proj,
                    const float* __restrict__ bias_table,
                    const unsigned short* __restrict__ wqkvT,
                    const unsigned short* __restrict__ wprojT,
                    float* __restrict__ out) {
  extern __shared__ char smem_raw[];
  unsigned short* Xs = (unsigned short*)smem_raw;          // [64][256] bf16 (reused as Os)
  unsigned short* Qs = Xs + 16384;                         // [64][256] bf16 (pre-scaled)
  unsigned short* Ks = Xs + 32768;                         // [64][256] bf16
  unsigned short* Vt = Xs + 49152;                         // [256][64] bf16 (V^T: [d][m])
  unsigned short* Ps = Xs + 65536;                         // [8][64][64] bf16 probs
  float*          Bt = (float*)(smem_raw + 196608);        // [127][8] rel-pos bias
  float*          Ms = (float*)(smem_raw + 212992);        // [64][64] window mask

  const int b    = blockIdx.x;
  const int widx = b & (NW - 1);
  const int tid  = threadIdx.x;
  // Wave id is uniform per wave: pin to SGPR so tile-selection is scalar-branched.
  const int w    = __builtin_amdgcn_readfirstlane(tid >> 5);
  const int lane = tid & 31;
  const int lm   = lane & 15;
  const int lg   = lane >> 4;

  // ---- stage X tile (fp32->bf16), mask tile and bias table into LDS ----
  {
    const float4* xg = (const float4*)(x + (size_t)b * WIN * DIM);
    unsigned int* xs32 = (unsigned int*)Xs;
    for (int i = tid; i < WIN * DIM / 4; i += 256) {
      const float4 f = xg[i];
      xs32[2 * i]     = pack2(f.x, f.y);
      xs32[2 * i + 1] = pack2(f.z, f.w);
    }
    const float4* mg = (const float4*)(mask + (size_t)widx * WIN * WIN);
    float4* ms4 = (float4*)Ms;
    for (int i = tid; i < WIN * WIN / 4; i += 256)
      ms4[i] = mg[i];
    for (int i = tid; i < (2 * WIN - 1) * HEADS; i += 256)
      Bt[i] = bias_table[i];
  }
  __syncthreads();

  // ---- Phase A: QKV[64,768] = X @ Wqkv + b_qkv (each wave: 6 N-tiles x 4 M-tiles) ----
  for (int i = 0; i < 6; ++i) {
    const int n0 = (w + 8 * i) * 16;                       // scalar
    for (int mt = 0; mt < 4; ++mt) {
      f32x8 acc = {};
      #pragma unroll
      for (int ks = 0; ks < 8; ++ks) {
        WOp A, B;
        load_a(Xs, mt * 16 + lm, DIM, ks * 32, lg, A);
        load_b(wqkvT, n0 + lm, DIM, ks * 32, lg, B);
        acc = wmma_bf16(A, B, acc);
      }
      const int ncol = n0 + lm;
      const float bias = b_qkv[ncol];
      if (n0 < 256) {                                      // Q (pre-scaled), scalar branch
        #pragma unroll
        for (int r = 0; r < 8; ++r) {
          const int m = mt * 16 + r + 8 * lg;
          Qs[m * DIM + ncol] = f2bf((acc[r] + bias) * QK_SCALE);
        }
      } else if (n0 < 512) {                               // K
        #pragma unroll
        for (int r = 0; r < 8; ++r) {
          const int m = mt * 16 + r + 8 * lg;
          Ks[m * DIM + (ncol - 256)] = f2bf(acc[r] + bias);
        }
      } else {                                             // V transposed: rows contiguous
        unsigned int* vrow =
            (unsigned int*)(Vt + (ncol - 512) * WIN + mt * 16 + 8 * lg);
        #pragma unroll
        for (int r = 0; r < 8; r += 2)
          vrow[r >> 1] = pack2(acc[r] + bias, acc[r + 1] + bias);
      }
    }
  }
  __syncthreads();

  // ---- Phase B: wave h owns head h. S = Q_h K_h^T + relbias + mask; softmax -> Ps ----
  const int h = w;
  for (int mt = 0; mt < 4; ++mt) {
    WOp A;
    load_a(Qs, mt * 16 + lm, DIM, h * HEAD_DIM, lg, A);    // head_dim = 32: one K-step
    f32x8 acc[4];
    #pragma unroll
    for (int nt = 0; nt < 4; ++nt) {
      WOp B;
      load_b(Ks, nt * 16 + lm, DIM, h * HEAD_DIM, lg, B);
      f32x8 z = {};
      acc[nt] = wmma_bf16(A, B, z);
    }
    // C layout: element (row = mt*16 + r + 8*lg, col = nt*16 + lm).
    #pragma unroll
    for (int r = 0; r < 8; ++r) {
      const int irow = mt * 16 + r + 8 * lg;
      float vals[4];
      #pragma unroll
      for (int nt = 0; nt < 4; ++nt) {
        const int j = nt * 16 + lm;
        vals[nt] = acc[nt][r]
                 + Bt[(irow - j + WIN - 1) * HEADS + h]
                 + Ms[irow * WIN + j];
      }
      float mx = fmaxf(fmaxf(vals[0], vals[1]), fmaxf(vals[2], vals[3]));
      #pragma unroll
      for (int s = 1; s < 16; s <<= 1) mx = fmaxf(mx, __shfl_xor(mx, s, 32));
      float e[4], sum = 0.f;
      #pragma unroll
      for (int nt = 0; nt < 4; ++nt) { e[nt] = __expf(vals[nt] - mx); sum += e[nt]; }
      #pragma unroll
      for (int s = 1; s < 16; s <<= 1) sum += __shfl_xor(sum, s, 32);
      const float inv = 1.0f / sum;
      #pragma unroll
      for (int nt = 0; nt < 4; ++nt)
        Ps[(h * WIN + irow) * WIN + nt * 16 + lm] = f2bf(e[nt] * inv);
    }
  }
  __syncthreads();

  // ---- Phase C: O_h[64,32] = P_h[64,64] @ V_h[64,32]; write into Os (reuses Xs) ----
  unsigned short* Os = Xs;
  for (int mt = 0; mt < 4; ++mt) {
    #pragma unroll
    for (int nd = 0; nd < 2; ++nd) {
      f32x8 acc = {};
      #pragma unroll
      for (int ks = 0; ks < 2; ++ks) {
        WOp A, B;
        load_a(Ps + h * WIN * WIN, mt * 16 + lm, WIN, ks * 32, lg, A);
        load_b(Vt, h * HEAD_DIM + nd * 16 + lm, WIN, ks * 32, lg, B);
        acc = wmma_bf16(A, B, acc);
      }
      #pragma unroll
      for (int r = 0; r < 8; ++r) {
        const int m = mt * 16 + r + 8 * lg;
        Os[m * DIM + h * HEAD_DIM + nd * 16 + lm] = f2bf(acc[r]);
      }
    }
  }
  __syncthreads();

  // ---- Phase D: Y[64,256] = O @ Wproj + b_proj -> global fp32 ----
  float* outb = out + (size_t)b * WIN * DIM;
  for (int i = 0; i < 2; ++i) {
    const int n0 = (w + 8 * i) * 16;                       // scalar
    for (int mt = 0; mt < 4; ++mt) {
      f32x8 acc = {};
      #pragma unroll
      for (int ks = 0; ks < 8; ++ks) {
        WOp A, B;
        load_a(Os, mt * 16 + lm, DIM, ks * 32, lg, A);
        load_b(wprojT, n0 + lm, DIM, ks * 32, lg, B);
        acc = wmma_bf16(A, B, acc);
      }
      const int ncol = n0 + lm;
      const float bias = b_proj[ncol];
      #pragma unroll
      for (int r = 0; r < 8; ++r) {
        const int m = mt * 16 + r + 8 * lg;
        outb[m * DIM + ncol] = acc[r] + bias;
      }
    }
  }
}

extern "C" void kernel_launch(void* const* d_in, const int* in_sizes, int n_in,
                              void* d_out, int out_size, void* d_ws, size_t ws_size,
                              hipStream_t stream) {
  const float* x          = (const float*)d_in[0];
  const float* mask       = (const float*)d_in[1];
  const float* w_qkv      = (const float*)d_in[2];
  const float* b_qkv      = (const float*)d_in[3];
  const float* w_proj     = (const float*)d_in[4];
  const float* b_proj     = (const float*)d_in[5];
  const float* bias_table = (const float*)d_in[6];

  unsigned short* wqkvT  = (unsigned short*)d_ws;              // [768][256] bf16
  unsigned short* wprojT = wqkvT + DIM * 3 * DIM;              // [256][256] bf16

  convert_weights<<<208, 256, 0, stream>>>(w_qkv, w_proj, wqkvT, wprojT);

  const int B_ = in_sizes[0] / (WIN * DIM);                    // 4096 windows
  const size_t lds_bytes = 229376;                             // 224 KB
  winattn_kernel<<<B_, 256, lds_bytes, stream>>>(x, mask, b_qkv, b_proj, bias_table,
                                                 wqkvT, wprojT, (float*)d_out);
}
